// DeformableTransformerDecoderLayer_3161095930122
// MI455X (gfx1250) — matrix operand order repaired and checked by reading, not compiled, not run on hardware
//
#include <hip/hip_runtime.h>
#include <hip/hip_bf16.h>
#include <cstdint>
#include <cstddef>

// ---------------------------------------------------------------------------
// Problem constants (from reference)
// ---------------------------------------------------------------------------
#define D_    256
#define NH_   8
#define HD_   32
#define NP_   8
#define LQ_   4096
#define NB_   8
#define GW_   80
#define HW_   (GW_*GW_)   // 6400
#define MLP_  1024
#define NOFF_ 128         // NH*NP*2
#define NAW_  64          // NH*NP
#define NOA_  192         // NOFF_+NAW_

typedef __attribute__((ext_vector_type(16))) __bf16 v16bf;
typedef __attribute__((ext_vector_type(8)))  float  v8f;

union BfFrag { v16bf v; uint4 u[2]; };

__device__ __forceinline__ unsigned short f2bf(float f) {
    unsigned int u = __builtin_bit_cast(unsigned int, f);
    u += 0x7FFFu + ((u >> 16) & 1u);          // round-to-nearest-even
    return (unsigned short)(u >> 16);
}
__device__ __forceinline__ float bf2f(unsigned short h) {
    unsigned int u = ((unsigned int)h) << 16;
    return __builtin_bit_cast(float, u);
}

// ---------------------------------------------------------------------------
// Weight repack: fp32 row-major (K x N) -> bf16 WMMA B-fragment layout.
// Fragment (kt,nt): 32 lanes x 16 elems, lane<16: n=ntBase+lane, k=ktBase+e;
// lane>=16: n=ntBase+(lane-16), k=ktBase+16+e.  Supports concat of two
// weight matrices along N (W2 may be null).
// ---------------------------------------------------------------------------
__global__ __launch_bounds__(256)
void repack_w_kernel(const float* __restrict__ W1, int N1,
                     const float* __restrict__ W2, int N2,
                     unsigned short* __restrict__ out, int K)
{
    const int N = N1 + N2;
    const size_t total = (size_t)K * N;
    size_t i = (size_t)blockIdx.x * 256 + threadIdx.x;
    if (i >= total) return;
    const int frag   = (int)(i >> 9);     // /512 (32 lanes * 16 elems)
    const int within = (int)(i & 511);
    const int lane   = within >> 4;
    const int e      = within & 15;
    const int NT     = N >> 4;
    const int kt     = frag / NT;
    const int nt     = frag - kt * NT;
    const int k      = kt * 32 + ((lane >> 4) << 4) + e;
    const int n      = nt * 16 + (lane & 15);
    float w = (n < N1) ? W1[(size_t)k * N1 + n]
                       : W2[(size_t)k * N2 + (n - N1)];
    out[i] = f2bf(w);
}

__global__ void concat_bias_kernel(const float* __restrict__ b1,
                                   const float* __restrict__ b2,
                                   float* __restrict__ out)
{
    int i = threadIdx.x;           // 192 threads
    out[i] = (i < NOFF_) ? b1[i] : b2[i - NOFF_];
}

// ---------------------------------------------------------------------------
// LayerNorm over last dim (256), fp32 in -> bf16 out. One wave per row.
// ---------------------------------------------------------------------------
__global__ __launch_bounds__(256)
void ln_kernel(const float* __restrict__ x,
               const float* __restrict__ g, const float* __restrict__ b,
               unsigned short* __restrict__ out, int rows)
{
    const int row  = blockIdx.x * 8 + (threadIdx.x >> 5);
    if (row >= rows) return;
    const int lane = threadIdx.x & 31;
    const float4* xr = (const float4*)(x + (size_t)row * D_);
    float4 v0 = xr[lane];
    float4 v1 = xr[lane + 32];
    float s  = v0.x + v0.y + v0.z + v0.w + v1.x + v1.y + v1.z + v1.w;
    float sq = v0.x*v0.x + v0.y*v0.y + v0.z*v0.z + v0.w*v0.w
             + v1.x*v1.x + v1.y*v1.y + v1.z*v1.z + v1.w*v1.w;
#pragma unroll
    for (int m = 16; m >= 1; m >>= 1) {
        s  += __shfl_xor(s,  m, 32);
        sq += __shfl_xor(sq, m, 32);
    }
    const float mean = s * (1.f / D_);
    const float inv  = rsqrtf(sq * (1.f / D_) - mean * mean + 1e-5f);
    const float4* gr = (const float4*)g;
    const float4* br = (const float4*)b;
    float4 g0 = gr[lane], g1 = gr[lane + 32];
    float4 b0 = br[lane], b1 = br[lane + 32];
    ushort4 o0, o1;
    o0.x = f2bf((v0.x - mean) * inv * g0.x + b0.x);
    o0.y = f2bf((v0.y - mean) * inv * g0.y + b0.y);
    o0.z = f2bf((v0.z - mean) * inv * g0.z + b0.z);
    o0.w = f2bf((v0.w - mean) * inv * g0.w + b0.w);
    o1.x = f2bf((v1.x - mean) * inv * g1.x + b1.x);
    o1.y = f2bf((v1.y - mean) * inv * g1.y + b1.y);
    o1.z = f2bf((v1.z - mean) * inv * g1.z + b1.z);
    o1.w = f2bf((v1.w - mean) * inv * g1.w + b1.w);
    ushort4* orow = (ushort4*)(out + (size_t)row * D_);
    orow[lane]      = o0;
    orow[lane + 32] = o1;
}

// ---------------------------------------------------------------------------
// WMMA bf16 GEMM:  C(MxN) = A(MxK, bf16 row-major) x Bpacked + bias [epilogue]
// Block: 256 threads = 8 waves, 128x64 block tile, 32x32 per wave.
// K, N multiples of 32/16; M multiple of 128.
// ---------------------------------------------------------------------------
template<bool GELU, bool RES, bool OUTBF16>
__global__ __launch_bounds__(256)
void gemm_bf16_wmma(const unsigned short* __restrict__ A,
                    const unsigned short* __restrict__ Bp,
                    const float* __restrict__ bias,
                    const float* __restrict__ resid,
                    void* __restrict__ outp,
                    int M, int N, int K)
{
    const int lane = threadIdx.x & 31;
    const int wave = threadIdx.x >> 5;
    const int wm   = wave & 3;            // 0..3
    const int wn   = wave >> 2;           // 0..1
    const int mW   = blockIdx.x * 128 + wm * 32;
    const int nW   = blockIdx.y * 64  + wn * 32;
    const int NT   = N >> 4;
    const int r    = lane & 15;
    const int koff = (lane >> 4) * 16;    // A sub-K offset per half-wave

    v8f acc00 = {0.f,0.f,0.f,0.f,0.f,0.f,0.f,0.f};
    v8f acc01 = acc00, acc10 = acc00, acc11 = acc00;

    const int KT  = K >> 5;
    const int nt0 = nW >> 4;

    for (int kt = 0; kt < KT; ++kt) {
        BfFrag a0, a1, b0, b1;
        const unsigned short* ar0 = A + (size_t)(mW + r)      * K + kt * 32 + koff;
        const unsigned short* ar1 = A + (size_t)(mW + 16 + r) * K + kt * 32 + koff;
        a0.u[0] = *(const uint4*)(ar0);
        a0.u[1] = *(const uint4*)(ar0 + 8);
        a1.u[0] = *(const uint4*)(ar1);
        a1.u[1] = *(const uint4*)(ar1 + 8);
        const unsigned short* bp0 = Bp + (((size_t)kt * NT + nt0)     * 32 + lane) * 16;
        const unsigned short* bp1 = Bp + (((size_t)kt * NT + nt0 + 1) * 32 + lane) * 16;
        b0.u[0] = *(const uint4*)(bp0);
        b0.u[1] = *(const uint4*)(bp0 + 8);
        b1.u[0] = *(const uint4*)(bp1);
        b1.u[1] = *(const uint4*)(bp1 + 8);

        acc00 = __builtin_amdgcn_wmma_f32_16x16x32_bf16(false, a0.v, false, b0.v,
                                                        (short)0, acc00, false, false);
        acc01 = __builtin_amdgcn_wmma_f32_16x16x32_bf16(false, a0.v, false, b1.v,
                                                        (short)0, acc01, false, false);
        acc10 = __builtin_amdgcn_wmma_f32_16x16x32_bf16(false, a1.v, false, b0.v,
                                                        (short)0, acc10, false, false);
        acc11 = __builtin_amdgcn_wmma_f32_16x16x32_bf16(false, a1.v, false, b1.v,
                                                        (short)0, acc11, false, false);
    }

    // Epilogue. C layout: lane<16 -> (M=j,     N=lane); lane>=16 -> (M=8+j, N=lane-16)
    const int mhalf = (lane >> 4) << 3;
    unsigned short* outb = (unsigned short*)outp;
    float*          outf = (float*)outp;

    auto store_tile = [&](v8f& acc, int mBase, int nBase) {
        const int col = nBase + r;
        const float bv = bias[col];
#pragma unroll
        for (int j = 0; j < 8; ++j) {
            const int row = mBase + mhalf + j;
            float v = acc[j] + bv;
            if (GELU) v = 0.5f * v * (1.f + erff(v * 0.70710678118654752f));
            if (RES)  v += resid[(size_t)row * N + col];
            if (OUTBF16) outb[(size_t)row * N + col] = f2bf(v);
            else         outf[(size_t)row * N + col] = v;
        }
    };
    store_tile(acc00, mW,      nW);
    store_tile(acc01, mW,      nW + 16);
    store_tile(acc10, mW + 16, nW);
    store_tile(acc11, mW + 16, nW + 16);
    (void)M;
}

// ---------------------------------------------------------------------------
// Deformable sampling. One block per (n,q), one wave per head, lane = dim.
// offaw row layout: [0:128) offsets (h*16 + p*2 + {x,y}), [128:192) aw logits.
// value: (NB, HW, NH*HD) bf16.  sampled: (NB*LQ, NH*HD) bf16.
// ---------------------------------------------------------------------------
__global__ __launch_bounds__(256)
void sample_kernel(const float* __restrict__ offaw,
                   const float* __restrict__ refp,
                   const unsigned short* __restrict__ value,
                   unsigned short* __restrict__ sampled)
{
    const int m = blockIdx.x;             // n*LQ + q
    const int h = threadIdx.x >> 5;
    const int d = threadIdx.x & 31;
    const int n = m >> 12;                // / LQ_ (4096)

    const float* oa = offaw + (size_t)m * NOA_;
    const float rx = refp[(size_t)m * 2 + 0];
    const float ry = refp[(size_t)m * 2 + 1];

    // softmax over the 8 attention logits of this head
    float lg[NP_];
    float mx = -3.0e38f;
#pragma unroll
    for (int p = 0; p < NP_; ++p) {
        lg[p] = oa[NOFF_ + h * NP_ + p];
        mx = fmaxf(mx, lg[p]);
    }
    float sum = 0.f;
#pragma unroll
    for (int p = 0; p < NP_; ++p) { lg[p] = __expf(lg[p] - mx); sum += lg[p]; }
    const float rs = 1.f / sum;

    const unsigned short* vbase =
        value + (size_t)n * HW_ * D_ + h * HD_ + d;

    float acc = 0.f;
#pragma unroll
    for (int p = 0; p < NP_; ++p) {
        const float ox = oa[h * 16 + p * 2 + 0];
        const float oy = oa[h * 16 + p * 2 + 1];
        // loc = ref + off/[W,H]; x = loc.x*W - 0.5  ==>  x = ref.x*W + off.x - 0.5
        const float x = rx * (float)GW_ + ox - 0.5f;
        const float y = ry * (float)GW_ + oy - 0.5f;
        const float x0f = floorf(x), y0f = floorf(y);
        const int   x0 = (int)x0f,  y0 = (int)y0f;
        const float fx = x - x0f,   fy = y - y0f;
        const float wp = lg[p] * rs;
#pragma unroll
        for (int dy = 0; dy < 2; ++dy) {
#pragma unroll
            for (int dx = 0; dx < 2; ++dx) {
                const int xc = x0 + dx, yc = y0 + dy;
                if (xc >= 0 && xc < GW_ && yc >= 0 && yc < GW_) {
                    const float w = (dx ? fx : 1.f - fx) * (dy ? fy : 1.f - fy);
                    const float val = bf2f(vbase[(size_t)(yc * GW_ + xc) * D_]);
                    acc += wp * w * val;
                }
            }
        }
    }
    sampled[(size_t)m * D_ + h * HD_ + d] = f2bf(acc);
}

// ---------------------------------------------------------------------------
// Host-side orchestration
// ---------------------------------------------------------------------------
extern "C" void kernel_launch(void* const* d_in, const int* in_sizes, int n_in,
                              void* d_out, int out_size, void* d_ws, size_t ws_size,
                              hipStream_t stream)
{
    (void)in_sizes; (void)n_in; (void)out_size; (void)ws_size;

    const float* tgt   = (const float*)d_in[0];
    const float* refp  = (const float*)d_in[1];
    const float* src   = (const float*)d_in[2];
    const float* n1g   = (const float*)d_in[5];
    const float* n1b   = (const float*)d_in[6];
    const float* so_w  = (const float*)d_in[7];
    const float* so_b  = (const float*)d_in[8];
    const float* aw_w  = (const float*)d_in[9];
    const float* aw_b  = (const float*)d_in[10];
    const float* vp_w  = (const float*)d_in[11];
    const float* vp_b  = (const float*)d_in[12];
    const float* op_w  = (const float*)d_in[13];
    const float* op_b  = (const float*)d_in[14];
    const float* n2g   = (const float*)d_in[15];
    const float* n2b   = (const float*)d_in[16];
    const float* fc1_w = (const float*)d_in[17];
    const float* fc1_b = (const float*)d_in[18];
    const float* fc2_w = (const float*)d_in[19];
    const float* fc2_b = (const float*)d_in[20];

    const int Mq = NB_ * LQ_;   // 32768 query rows
    const int Ms = NB_ * HW_;   // 51200 source rows

    char* ws = (char*)d_ws;
    size_t cur = 0;
    auto alloc = [&](size_t bytes) -> void* {
        void* p = ws + cur;
        cur += (bytes + 255) & ~(size_t)255;
        return p;
    };
    unsigned short* q_bf     = (unsigned short*)alloc((size_t)Mq * D_ * 2);
    unsigned short* s_bf     = (unsigned short*)alloc((size_t)Ms * D_ * 2);
    unsigned short* value_bf = (unsigned short*)alloc((size_t)Ms * D_ * 2);
    float*          offaw    = (float*)         alloc((size_t)Mq * NOA_ * 4);
    unsigned short* samp_bf  = (unsigned short*)alloc((size_t)Mq * D_ * 2);
    float*          tgt2     = (float*)         alloc((size_t)Mq * D_ * 4);
    unsigned short* h_bf     = (unsigned short*)alloc((size_t)Mq * D_ * 2);
    unsigned short* h1_bf    = (unsigned short*)alloc((size_t)Mq * MLP_ * 2);
    unsigned short* pw_vp    = (unsigned short*)alloc((size_t)D_ * D_ * 2);
    unsigned short* pw_soaw  = (unsigned short*)alloc((size_t)D_ * NOA_ * 2);
    unsigned short* pw_op    = (unsigned short*)alloc((size_t)D_ * D_ * 2);
    unsigned short* pw_fc1   = (unsigned short*)alloc((size_t)D_ * MLP_ * 2);
    unsigned short* pw_fc2   = (unsigned short*)alloc((size_t)MLP_ * D_ * 2);
    float*          b_soaw   = (float*)         alloc(NOA_ * 4);

    // 1) weight repack + bias concat
    auto rp_blocks = [](size_t elems) { return (unsigned)((elems + 255) / 256); };
    repack_w_kernel<<<rp_blocks((size_t)D_ * D_),   256, 0, stream>>>(vp_w,  D_,    nullptr, 0,    pw_vp,   D_);
    repack_w_kernel<<<rp_blocks((size_t)D_ * NOA_), 256, 0, stream>>>(so_w,  NOFF_, aw_w,    NAW_, pw_soaw, D_);
    repack_w_kernel<<<rp_blocks((size_t)D_ * D_),   256, 0, stream>>>(op_w,  D_,    nullptr, 0,    pw_op,   D_);
    repack_w_kernel<<<rp_blocks((size_t)D_ * MLP_), 256, 0, stream>>>(fc1_w, MLP_,  nullptr, 0,    pw_fc1,  D_);
    repack_w_kernel<<<rp_blocks((size_t)MLP_ * D_), 256, 0, stream>>>(fc2_w, D_,    nullptr, 0,    pw_fc2,  MLP_);
    concat_bias_kernel<<<1, NOA_, 0, stream>>>(so_b, aw_b, b_soaw);

    // 2) LayerNorm1 on tgt -> q_bf and src -> s_bf
    ln_kernel<<<Mq / 8, 256, 0, stream>>>(tgt, n1g, n1b, q_bf, Mq);
    ln_kernel<<<Ms / 8, 256, 0, stream>>>(src, n1g, n1b, s_bf, Ms);

    // 3) value = s @ vp_w + vp_b        (51200 x 256 x 256) -> bf16
    gemm_bf16_wmma<false, false, true><<<dim3(Ms / 128, D_ / 64), 256, 0, stream>>>(
        s_bf, pw_vp, vp_b, nullptr, value_bf, Ms, D_, D_);

    // 4) offaw = q @ [so_w|aw_w] + bias (32768 x 256 x 192) -> f32
    gemm_bf16_wmma<false, false, false><<<dim3(Mq / 128, NOA_ / 64), 256, 0, stream>>>(
        q_bf, pw_soaw, b_soaw, nullptr, offaw, Mq, NOA_, D_);

    // 5) deformable bilinear sampling + softmax weights -> samp_bf
    sample_kernel<<<Mq, 256, 0, stream>>>(offaw, refp, value_bf, samp_bf);

    // 6) tgt2 = tgt + samp @ op_w + op_b   -> f32
    gemm_bf16_wmma<false, true, false><<<dim3(Mq / 128, D_ / 64), 256, 0, stream>>>(
        samp_bf, pw_op, op_b, tgt, tgt2, Mq, D_, D_);

    // 7) LayerNorm2 on tgt2 -> h_bf
    ln_kernel<<<Mq / 8, 256, 0, stream>>>(tgt2, n2g, n2b, h_bf, Mq);

    // 8) h1 = gelu(h @ fc1_w + fc1_b)     (32768 x 1024 x 256) -> bf16
    gemm_bf16_wmma<true, false, true><<<dim3(Mq / 128, MLP_ / 64), 256, 0, stream>>>(
        h_bf, pw_fc1, fc1_b, nullptr, h1_bf, Mq, MLP_, D_);

    // 9) out = tgt2 + h1 @ fc2_w + fc2_b  (32768 x 256 x 1024) -> f32 d_out
    gemm_bf16_wmma<false, true, false><<<dim3(Mq / 128, D_ / 64), 256, 0, stream>>>(
        h1_bf, pw_fc2, fc2_b, tgt2, d_out, Mq, D_, MLP_);
}
